// CrossAttentionBlock_26877905339233
// MI455X (gfx1250) — compile-verified, matrix operand order checked by reading, not run on hardware
//
#include <hip/hip_runtime.h>

// ---------------- problem constants ----------------
constexpr int kB   = 4;
constexpr int kC   = 256;
constexpr int kH   = 64;
constexpr int kW   = 64;
constexpr int kHW  = kH * kW;        // 4096
constexpr int kG   = 32;             // groups
constexpr int kCPG = kC / kG;        // 8 channels per group

// ---------------- vector types ----------------
typedef __attribute__((ext_vector_type(16))) __bf16 v16bf;
typedef __attribute__((ext_vector_type(8)))  float  v8f;
typedef __attribute__((ext_vector_type(4)))  float  f32x4;
typedef __attribute__((ext_vector_type(4)))  unsigned int v4u;
typedef __attribute__((ext_vector_type(8)))  int    v8i;
typedef __attribute__((ext_vector_type(4)))  int    v4i;

union Frag { v16bf v; f32x4 f[2]; };

__device__ __forceinline__ v8f wmma_bf16(v16bf a, v16bf b, v8f c) {
  return __builtin_amdgcn_wmma_f32_16x16x32_bf16(false, a, false, b, (short)0, c,
                                                 false, false);
}

__device__ __forceinline__ unsigned short f2bf(float f) {
  union { float f; unsigned u; } x; x.f = f;
  unsigned r = x.u + 0x7fffu + ((x.u >> 16) & 1u);   // round-to-nearest-even
  return (unsigned short)(r >> 16);
}

// Load a 16x32 bf16 A/B fragment from row-major [rows][rowstride] bf16 storage.
// Per lane: two contiguous 16B chunks at K offsets half*8 and 16+half*8
// (CDNA5 16-bit 16x32 matrix VGPR layout, ISA 7.12.2). Works for global or LDS.
__device__ __forceinline__ v16bf load_frag(const unsigned short* base, int rowstride,
                                           int row, int k0) {
  const char* p = (const char*)(base + (size_t)row * rowstride + k0);
  int half = (threadIdx.x >> 4) & 1;
  Frag fr;
  fr.f[0] = *(const f32x4*)(p + half * 16);
  fr.f[1] = *(const f32x4*)(p + 32 + half * 16);
  return fr.v;
}

// ---------------- TDM: 2-D tile load, global -> LDS ----------------
// Builds a D# (ISA 08_async_tensor §8): group0 = {count=1, lds_addr,
// global_addr[56:0], type=2}; group1 = {data_size=2B, tensor_dim0/1,
// tile_dim0/1, tensor_dim0_stride}. Groups 2/3 zero (<=2-D tensor).
// All dims/strides in elements (bf16).
__device__ __forceinline__ void tdm_load_2d(unsigned lds_off, const void* gptr,
                                            unsigned tensor_d0, unsigned tensor_d1,
                                            unsigned tile_d0, unsigned tile_d1,
                                            unsigned long long stride0) {
  unsigned long long ga = (unsigned long long)(uintptr_t)gptr;
  v4u g0;
  g0[0] = 1u;                                              // count=1, user mode
  g0[1] = lds_off;                                         // lds_addr (bytes)
  g0[2] = (unsigned)ga;                                    // global_addr[31:0]
  g0[3] = (unsigned)((ga >> 32) & 0x01FFFFFFu) | 0x80000000u; // addr[56:32] | type=2
  unsigned long long q0 = (1ull << 16)                                   // data_size=2B
                        | ((unsigned long long)(tensor_d0 & 0xFFFFu) << 48);
  unsigned long long q1 = ((unsigned long long)(tensor_d0 >> 16) & 0xFFFFull)
                        | ((unsigned long long)tensor_d1 << 16)
                        | ((unsigned long long)(tile_d0 & 0xFFFFu) << 48);
  unsigned long long q2 = (unsigned long long)(tile_d1 & 0xFFFFu)        // tile_dim1
                        | ((stride0 & 0xFFFFFFFFull) << 32);             // stride0 lo
  unsigned long long q3 = (stride0 >> 32) & 0xFFFFull;                   // stride0 hi
  v8i g1;
  g1[0] = (int)(unsigned)q0; g1[1] = (int)(unsigned)(q0 >> 32);
  g1[2] = (int)(unsigned)q1; g1[3] = (int)(unsigned)(q1 >> 32);
  g1[4] = (int)(unsigned)q2; g1[5] = (int)(unsigned)(q2 >> 32);
  g1[6] = (int)(unsigned)q3; g1[7] = (int)(unsigned)(q3 >> 32);
  v4i z4 = {0, 0, 0, 0};
#if defined(__clang_major__) && (__clang_major__ >= 23)
  v8i z8 = {0, 0, 0, 0, 0, 0, 0, 0};
  __builtin_amdgcn_tensor_load_to_lds(g0, g1, z4, z4, z8, 0);   // clang-23 arity
#else
  __builtin_amdgcn_tensor_load_to_lds(g0, g1, z4, z4, 0);       // ROCm 7.2 arity
#endif
}

// ---------------- 1) GroupNorm statistics ----------------
__global__ void k_gn_stats(const float* __restrict__ q, const float* __restrict__ kv,
                           float* __restrict__ stats) {
  int g = blockIdx.x, b = blockIdx.y, which = blockIdx.z;
  const float* x = (which == 0 ? q : kv) + ((size_t)b * kC + g * kCPG) * kHW;
  const int N = kCPG * kHW;
  float s = 0.f, s2 = 0.f;
  for (int i = threadIdx.x; i < N; i += 256) { float v = x[i]; s += v; s2 += v * v; }
  __shared__ float sh0[256], sh1[256];
  sh0[threadIdx.x] = s; sh1[threadIdx.x] = s2;
  __syncthreads();
  for (int off = 128; off > 0; off >>= 1) {
    if (threadIdx.x < off) { sh0[threadIdx.x] += sh0[threadIdx.x + off];
                             sh1[threadIdx.x] += sh1[threadIdx.x + off]; }
    __syncthreads();
  }
  if (threadIdx.x == 0) {
    float mean = sh0[0] / N;
    float var  = sh1[0] / N - mean * mean;
    int idx = ((which * kB) + b) * kG + g;
    stats[idx * 2 + 0] = mean;
    stats[idx * 2 + 1] = rsqrtf(var + 1e-5f);
  }
}

// ---------------- 2) GroupNorm apply + cast + transpose to [b][hw][c] ----------------
__global__ void k_gn_apply(const float* __restrict__ q, const float* __restrict__ kv,
                           const float* __restrict__ gw, const float* __restrict__ gb,
                           const float* __restrict__ stats,
                           unsigned short* __restrict__ qnT,
                           unsigned short* __restrict__ kvnT) {
  int hw = blockIdx.x, b = blockIdx.y, which = blockIdx.z;
  int c = threadIdx.x;
  const float* x = which ? kv : q;
  float v = x[((size_t)b * kC + c) * kHW + hw];
  int sidx = (((which * kB) + b) * kG + (c / kCPG)) * 2;
  float y = (v - stats[sidx]) * stats[sidx + 1] * gw[c] + gb[c];
  unsigned short* o = which ? kvnT : qnT;
  o[((size_t)b * kHW + hw) * kC + c] = f2bf(y);
}

// ---------------- 3) Weight casts ----------------
__global__ void k_cast_w(const float* __restrict__ wq, const float* __restrict__ wkv,
                         const float* __restrict__ wo,
                         unsigned short* __restrict__ wq_b,
                         unsigned short* __restrict__ wkv_b,
                         unsigned short* __restrict__ woT_b) {
  int i = blockIdx.x * 256 + threadIdx.x;
  if (i < kC * kC)     wq_b[i]  = f2bf(wq[i]);
  if (i < 2 * kC * kC) wkv_b[i] = f2bf(wkv[i]);
  if (i < 9 * kC * kC) {
    int tap = i / (kC * kC), rem = i % (kC * kC);
    int o = rem / kC, c = rem % kC;
    woT_b[i] = f2bf(wo[((size_t)o * kC + c) * 9 + tap]);
  }
}

// ---------------- 4) Projections (WMMA GEMM) ----------------
__global__ __launch_bounds__(256)
void k_project(const unsigned short* __restrict__ qnT, const unsigned short* __restrict__ kvnT,
               const unsigned short* __restrict__ wq_b, const unsigned short* __restrict__ wkv_b,
               const float* __restrict__ bq, const float* __restrict__ bkv,
               unsigned short* __restrict__ qpT, unsigned short* __restrict__ kpT,
               unsigned short* __restrict__ vp) {
  int wave = (blockIdx.x * 256 + threadIdx.x) >> 5;
  int lane = threadIdx.x & 31, sel = lane & 15, half = lane >> 4;
  int nt = wave % 48;
  int mt = wave / 48;
  int b  = mt >> 8;
  int q0 = (mt & 255) * 16;
  bool is_q = (nt < 16);
  const unsigned short* X  = (is_q ? qnT : kvnT) + (size_t)b * kHW * kC;
  const unsigned short* Wm = is_q ? wq_b : wkv_b;
  int o0 = (is_q ? nt : nt - 16) * 16;
  v8f acc = {0, 0, 0, 0, 0, 0, 0, 0};
#pragma unroll
  for (int cc = 0; cc < 8; cc++) {
    v16bf a = load_frag(X, kC, q0 + sel, cc * 32);
    v16bf w = load_frag(Wm, kC, o0 + sel, cc * 32);
    acc = wmma_bf16(a, w, acc);
  }
  if (is_q) {
    float bias = bq[o0 + sel];
#pragma unroll
    for (int r = 0; r < 8; r++) {
      int m = r + half * 8;
      qpT[((size_t)b * kHW + q0 + m) * kC + o0 + sel] = f2bf(acc[r] + bias);
    }
  } else {
    int oc = o0 + sel;
    float bias = bkv[oc];
    int c_out = oc >> 1, j = oc & 1;
#pragma unroll
    for (int r = 0; r < 8; r++) {
      int m = r + half * 8;
      float v = acc[r] + bias;
      if (j == 0)
        kpT[((size_t)b * kHW + q0 + m) * kC + c_out] = f2bf(v * 0.0625f);
      else
        vp[((size_t)b * kC + c_out) * kHW + q0 + m] = f2bf(v);
    }
  }
}

// ---------------- 5) Flash attention with TDM double buffering ----------------
// Block = 4 waves, 64 queries. Wave 0 drives the Tensor Data Mover: per 32-key
// tile it DMAs K (32x256, [key][c]) and V (256x32, [c][key]) into ping-pong LDS
// buffers (TENSORcnt / s_wait_tensorcnt), overlapping the DMA of tile i+1 with
// the 32 WMMAs + online softmax of tile i.
__global__ __launch_bounds__(128)
void k_attn(const unsigned short* __restrict__ qpT, const unsigned short* __restrict__ kpT,
            const unsigned short* __restrict__ vp, unsigned short* __restrict__ aT) {
  __shared__ unsigned short ldsK[2][32 * kC];     // 2 x 16 KB
  __shared__ unsigned short ldsV[2][kC * 32];     // 2 x 16 KB
  __shared__ unsigned short ldsP[4 * 16 * 32];    // per-wave P staging, 4 KB
  int wid = threadIdx.x >> 5, lane = threadIdx.x & 31;
  int sel = lane & 15, half = lane >> 4;
  int b  = blockIdx.x >> 6;
  int q0 = (blockIdx.x & 63) * 64 + wid * 16;
  const unsigned short* Q = qpT + (size_t)b * kHW * kC;
  const unsigned short* K = kpT + (size_t)b * kHW * kC;
  const unsigned short* V = vp  + (size_t)b * kC * kHW;
  unsigned short* myP = ldsP + wid * 512;

  auto issue = [&](int i) {
    int buf = i & 1;
    // K tile: dim0 = c (256 contiguous), dim1 = key rows, 32 rows starting at key i*32
    tdm_load_2d((unsigned)(uintptr_t)&ldsK[buf][0], K + (size_t)i * 32 * kC,
                kC, kHW, kC, 32, kC);
    // V tile: dim0 = key (32 contiguous starting at i*32), dim1 = all 256 channels
    tdm_load_2d((unsigned)(uintptr_t)&ldsV[buf][0], V + i * 32,
                kHW, kC, 32, kC, kHW);
  };

  v16bf afr[8];
#pragma unroll
  for (int cc = 0; cc < 8; cc++) afr[cc] = load_frag(Q, kC, q0 + sel, cc * 32);

  v8f acc[16];
#pragma unroll
  for (int ct = 0; ct < 16; ct++) acc[ct] = (v8f){0, 0, 0, 0, 0, 0, 0, 0};
  float mrun[8], lrun[8];
#pragma unroll
  for (int r = 0; r < 8; r++) { mrun[r] = -1e30f; lrun[r] = 0.f; }

  constexpr int nTiles = kHW / 32;                // 128
  if (wid == 0) issue(0);
  for (int i = 0; i < nTiles; i++) {
    if (wid == 0) __builtin_amdgcn_s_wait_tensorcnt(0);   // tile i landed
    __syncthreads();                                      // publish LDS to all waves
    if (wid == 0 && i + 1 < nTiles) issue(i + 1);         // prefetch next tile
    const unsigned short* Kb = &ldsK[i & 1][0];
    const unsigned short* Vb = &ldsV[i & 1][0];

    v8f s0 = {0, 0, 0, 0, 0, 0, 0, 0}, s1 = {0, 0, 0, 0, 0, 0, 0, 0};
#pragma unroll
    for (int cc = 0; cc < 8; cc++) {
      v16bf b0 = load_frag(Kb, kC, sel,      cc * 32);
      v16bf b1 = load_frag(Kb, kC, 16 + sel, cc * 32);
      s0 = wmma_bf16(afr[cc], b0, s0);
      s1 = wmma_bf16(afr[cc], b1, s1);
    }
#pragma unroll
    for (int r = 0; r < 8; r++) {
      float rm = fmaxf(s0[r], s1[r]);
      rm = fmaxf(rm, __shfl_xor(rm, 1));
      rm = fmaxf(rm, __shfl_xor(rm, 2));
      rm = fmaxf(rm, __shfl_xor(rm, 4));
      rm = fmaxf(rm, __shfl_xor(rm, 8));
      float mnew  = fmaxf(mrun[r], rm);
      float alpha = __expf(mrun[r] - mnew);
      mrun[r] = mnew;
      float p0 = __expf(s0[r] - mnew);
      float p1 = __expf(s1[r] - mnew);
      float rs = p0 + p1;
      rs += __shfl_xor(rs, 1);
      rs += __shfl_xor(rs, 2);
      rs += __shfl_xor(rs, 4);
      rs += __shfl_xor(rs, 8);
      lrun[r] = lrun[r] * alpha + rs;
#pragma unroll
      for (int ct = 0; ct < 16; ct++) acc[ct][r] *= alpha;
      int m = r + half * 8;
      myP[m * 32 + sel]      = f2bf(p0);
      myP[m * 32 + 16 + sel] = f2bf(p1);
    }
    __builtin_amdgcn_wave_barrier();
    Frag pf;
    pf.f[0] = *(const f32x4*)((const char*)myP + sel * 64 + half * 16);
    pf.f[1] = *(const f32x4*)((const char*)myP + sel * 64 + 32 + half * 16);
#pragma unroll
    for (int ct = 0; ct < 16; ct++) {
      v16bf vb = load_frag(Vb, 32, ct * 16 + sel, 0);
      acc[ct] = wmma_bf16(pf.v, vb, acc[ct]);
    }
    __syncthreads();                 // all waves done with buf[i&1] before reuse
  }
#pragma unroll
  for (int r = 0; r < 8; r++) {
    float inv = 1.0f / lrun[r];
    int m = r + half * 8;
#pragma unroll
    for (int ct = 0; ct < 16; ct++)
      aT[((size_t)b * kHW + q0 + m) * kC + ct * 16 + sel] = f2bf(acc[ct][r] * inv);
  }
}

// ---------------- 6) 3x3 conv (implicit GEMM) + bias + residual ----------------
__global__ __launch_bounds__(256)
void k_conv(const unsigned short* __restrict__ aT, const unsigned short* __restrict__ woT_b,
            const float* __restrict__ bo, const float* __restrict__ qin,
            float* __restrict__ out) {
  int wave = (blockIdx.x * 256 + threadIdx.x) >> 5;
  int lane = threadIdx.x & 31, sel = lane & 15, half = lane >> 4;
  int ot = wave & 15;
  int pt = wave >> 4;
  int b  = pt >> 8;
  int p0 = (pt & 255) * 16;
  int h  = p0 >> 6, w0 = p0 & 63;
  const unsigned short* A = aT + (size_t)b * kHW * kC;
  int o0 = ot * 16;
  v8f acc = {0, 0, 0, 0, 0, 0, 0, 0};
#pragma unroll
  for (int tap = 0; tap < 9; tap++) {
    int dy = tap / 3 - 1, dx = tap % 3 - 1;
    int hh = h + dy;
    if (hh < 0 || hh >= kH) continue;
    int wc = w0 + sel + dx;
    bool ok = (wc >= 0 && wc < kW);
    int arow = hh * kW + (ok ? wc : 0);
    const unsigned short* Wt = woT_b + (size_t)tap * kC * kC;
#pragma unroll
    for (int cc = 0; cc < 8; cc++) {
      v16bf av = load_frag(A, kC, arow, cc * 32);
      if (!ok) { Frag z; z.f[0] = (f32x4){0,0,0,0}; z.f[1] = (f32x4){0,0,0,0}; av = z.v; }
      v16bf wv = load_frag(Wt, kC, o0 + sel, cc * 32);
      acc = wmma_bf16(av, wv, acc);
    }
  }
  float bias = bo[o0 + sel];
#pragma unroll
  for (int r = 0; r < 8; r++) {
    int m = r + half * 8;
    size_t oidx = ((size_t)b * kC + o0 + sel) * kHW + p0 + m;
    out[oidx] = acc[r] + bias + qin[oidx];
  }
}

// ---------------- launch ----------------
extern "C" void kernel_launch(void* const* d_in, const int* in_sizes, int n_in,
                              void* d_out, int out_size, void* d_ws, size_t ws_size,
                              hipStream_t stream) {
  (void)in_sizes; (void)n_in; (void)out_size; (void)ws_size;
  const float* q   = (const float*)d_in[0];
  const float* kv  = (const float*)d_in[1];
  const float* gw  = (const float*)d_in[2];
  const float* gb  = (const float*)d_in[3];
  const float* wq  = (const float*)d_in[4];
  const float* bq  = (const float*)d_in[5];
  const float* wkv = (const float*)d_in[6];
  const float* bkv = (const float*)d_in[7];
  const float* wo  = (const float*)d_in[8];
  const float* bo  = (const float*)d_in[9];
  float* out = (float*)d_out;

  char* ws = (char*)d_ws;
  size_t off = 0;
  auto alloc = [&](size_t bytes) -> void* {
    void* p = ws + off;
    off = (off + bytes + 255) & ~(size_t)255;
    return p;
  };
  const size_t bfTensor = (size_t)kB * kHW * kC * 2;       // 8 MB
  float*          stats  = (float*)alloc(2 * kB * kG * 2 * sizeof(float));
  unsigned short* qnT    = (unsigned short*)alloc(bfTensor);
  unsigned short* kvnT   = (unsigned short*)alloc(bfTensor);
  unsigned short* qpT    = (unsigned short*)alloc(bfTensor);
  unsigned short* kpT    = (unsigned short*)alloc(bfTensor);
  unsigned short* vp     = (unsigned short*)alloc(bfTensor);
  unsigned short* wq_b   = (unsigned short*)alloc((size_t)kC * kC * 2);
  unsigned short* wkv_b  = (unsigned short*)alloc((size_t)2 * kC * kC * 2);
  unsigned short* woT_b  = (unsigned short*)alloc((size_t)9 * kC * kC * 2);
  unsigned short* aT     = qnT;   // qnT dead after k_project; reuse

  k_gn_stats<<<dim3(kG, kB, 2), 256, 0, stream>>>(q, kv, stats);
  k_gn_apply<<<dim3(kHW, kB, 2), 256, 0, stream>>>(q, kv, gw, gb, stats, qnT, kvnT);
  k_cast_w<<<(9 * kC * kC + 255) / 256, 256, 0, stream>>>(wq, wkv, wo, wq_b, wkv_b, woT_b);
  k_project<<<49152 / 8, 256, 0, stream>>>(qnT, kvnT, wq_b, wkv_b, bq, bkv, qpT, kpT, vp);
  k_attn<<<kB * 64, 128, 0, stream>>>(qpT, kpT, vp, aT);
  k_conv<<<16384 / 8, 256, 0, stream>>>(aT, woT_b, bo, q, out);
}
